// BLSTM_GAT_CRF_21165598834733
// MI455X (gfx1250) — matrix operand order repaired and checked by reading, not compiled, not run on hardware
//
#include <hip/hip_runtime.h>
#include <cstdint>
#include <cstddef>

typedef __attribute__((ext_vector_type(16))) __bf16 v16bf;
typedef __attribute__((ext_vector_type(8)))  __bf16 v8bf;
typedef __attribute__((ext_vector_type(8)))  float  v8f;

#define DEV __device__ __forceinline__

DEV v8f wmma_bf16(v16bf a, v16bf b, v8f c) {
  return __builtin_amdgcn_wmma_f32_16x16x32_bf16(false, a, false, b, (short)0, c, false, false);
}
DEV v8f vzero8() { v8f z = {0.f,0.f,0.f,0.f,0.f,0.f,0.f,0.f}; return z; }
DEV __bf16 f2bf(float x) { return (__bf16)x; }
DEV float bf2f(__bf16 x) { return (float)x; }

struct bfx16 { v8bf lo, hi; };
DEV v16bf join16(v8bf lo, v8bf hi) { bfx16 t{lo, hi}; return __builtin_bit_cast(v16bf, t); }

// ---- fast activations (hardware transcendentals, no IEEE-div expansion) ----
DEV float fast_rcp(float x) { return __builtin_amdgcn_rcpf(x); }
DEV float fast_sigmoid(float x) { return fast_rcp(1.f + __expf(-x)); }
#if __has_builtin(__builtin_amdgcn_tanhf)
DEV float fast_tanh(float x) { return __builtin_amdgcn_tanhf(x); }
#elif __has_builtin(__builtin_amdgcn_tanh_f32)
DEV float fast_tanh(float x) { return __builtin_amdgcn_tanh_f32(x); }
#else
DEV float fast_tanh(float x) {
  float xc = fminf(fmaxf(x, -15.f), 15.f);
  float e = __expf(2.f * xc);
  return (e - 1.f) * fast_rcp(e + 1.f);
}
#endif

// ---- gfx1250 async global->LDS copies (ASYNCcnt-tracked; LDS addr = flat[31:0]) ----
DEV unsigned lds_addr32(const void* p) { return (unsigned)(uintptr_t)p; }
DEV void async_copy_b128(unsigned ldsAddr, const void* gptr) {
  asm volatile("global_load_async_to_lds_b128 %0, %1, off"
               :: "v"(ldsAddr), "v"((unsigned long long)(uintptr_t)gptr)
               : "memory");
}
DEV void async_copy_b32(unsigned ldsAddr, const void* gptr) {
  asm volatile("global_load_async_to_lds_b32 %0, %1, off"
               :: "v"(ldsAddr), "v"((unsigned long long)(uintptr_t)gptr)
               : "memory");
}
DEV void async_wait0() {
  asm volatile("s_wait_asynccnt 0" ::: "memory");
}

// ---------------- constants ----------------
#define Bc   16
#define Sc   384
#define Gc   384
#define Nc   768
#define Tc   21
#define STARTc 19
#define STOPc  20
#define AW   24            // adjacency bitmask words per row (768/32)

// ---------------- weight prep ----------------
__global__ void prep_kernel(const float* __restrict__ wif, const float* __restrict__ wib,
                            const float* __restrict__ whf, const float* __restrict__ whb,
                            const float* __restrict__ h2hW, const float* __restrict__ h2hb,
                            const float* __restrict__ gWh,  const float* __restrict__ gWo,
                            __bf16* wihT, __bf16* whhT, __bf16* WhT, __bf16* WoT,
                            __bf16* h2hT, float* h2hbP, float* zbias) {
  int idx = blockIdx.x * blockDim.x + threadIdx.x;
  const int S0 = 2*128*256;
  const int S1 = S0 + 2*64*256;
  const int S2 = S1 + 12*128*64;
  const int S3 = S2 + 3*256*32;
  const int S4 = S3 + 128*32;
  const int S5 = S4 + 32;
  const int S6 = S5 + 256;
  if (idx < S0) {
    int dir = idx >> 15, rem = idx & 32767;
    int k = rem >> 8, g = rem & 255;
    const float* w = dir ? wib : wif;
    wihT[idx] = f2bf((k < 100) ? w[g*100 + k] : 0.f);
  } else if (idx < S1) {
    int r = idx - S0;
    int dir = r >> 14, d = r & 16383;
    int tileIdx = d >> 9;
    int lane = (d >> 4) & 31, e = d & 15;
    int ks = tileIdx & 1, nt = (tileIdx >> 1) & 1, w = tileIdx >> 2;
    int k = ks*32 + (lane >> 4)*16 + e;
    int g = w*32 + nt*16 + (lane & 15);
    const float* wsrc = dir ? whb : whf;
    whhT[r] = f2bf(wsrc[g*64 + k]);
  } else if (idx < S2) {
    WhT[idx - S1] = f2bf(gWh[idx - S1]);
  } else if (idx < S3) {
    int r = idx - S2;
    int g = r >> 13, rem = r & 8191;
    int k = rem >> 5, n = rem & 31;
    WoT[r] = f2bf((n < Tc) ? gWo[(g*256 + k)*Tc + n] : 0.f);
  } else if (idx < S4) {
    int r = idx - S3;
    int k = r >> 5, n = r & 31;
    h2hT[r] = f2bf((n < Tc) ? h2hW[n*128 + k] : 0.f);
  } else if (idx < S5) {
    int n = idx - S4;
    h2hbP[n] = (n < Tc) ? h2hb[n] : 0.f;
  } else if (idx < S6) {
    zbias[idx - S5] = 0.f;
  }
}

// ---------------- adjacency bit-pack: [g][b][i][AW] ----------------
__global__ void packadj_kernel(const int* __restrict__ g0, const int* __restrict__ g1,
                               const int* __restrict__ g2, unsigned* __restrict__ out,
                               int wordsPerGraph) {
  int idx = blockIdx.x * blockDim.x + threadIdx.x;
  if (idx >= 3*wordsPerGraph) return;
  int g = idx / wordsPerGraph;
  int w = idx % wordsPerGraph;
  const int* adj = (g == 0) ? g0 : (g == 1) ? g1 : g2;
  int row = w / AW, word = w % AW;
  const int4* src = (const int4*)(adj + (size_t)row*Nc + word*32);
  unsigned bits = 0;
#pragma unroll
  for (int q = 0; q < 8; ++q) {
    int4 v = src[q];
    if (v.x > 0) bits |= 1u << (q*4 + 0);
    if (v.y > 0) bits |= 1u << (q*4 + 1);
    if (v.z > 0) bits |= 1u << (q*4 + 2);
    if (v.w > 0) bits |= 1u << (q*4 + 3);
  }
  out[idx] = bits;
}

// ---------------- gathers ----------------
__global__ void embed_kernel(const int* __restrict__ chars, const float* __restrict__ table,
                             __bf16* __restrict__ emb, int total) {
  int idx = blockIdx.x * blockDim.x + threadIdx.x;
  if (idx >= total) return;
  int i = idx >> 7, k = idx & 127;
  int c = chars[i];
  emb[idx] = f2bf((k < 100) ? table[(size_t)c*100 + k] : 0.f);
}

__global__ void gaz_kernel(const int* __restrict__ gaz, const float* __restrict__ table,
                           __bf16* __restrict__ gat_in, int total) {
  int idx = blockIdx.x * blockDim.x + threadIdx.x;
  if (idx >= total) return;
  int bj = idx >> 7, d = idx & 127;
  int b = bj / Gc, j = bj % Gc;
  int g = gaz[bj];
  gat_in[((size_t)(b*Nc + Sc + j))*128 + d] = f2bf(table[(size_t)g*128 + d]);
}

// ---------------- generic WMMA GEMM ----------------
template<bool CBF, bool RSWAP>
__global__ __launch_bounds__(256) void gemm_kernel(
    const __bf16* __restrict__ A, int lda, int rowMod, int rowStride,
    const __bf16* __restrict__ Bm, int ldb,
    void* __restrict__ C, int ldc,
    const float* __restrict__ bias, int K) {
  __shared__ __attribute__((aligned(32))) __bf16 As[128*32];
  __shared__ __attribute__((aligned(32))) __bf16 Bs[2*32*16];
  int tid = threadIdx.x;
  int wave = tid >> 5, lane = tid & 31;
  int m0 = blockIdx.x * 128;
  int n0 = blockIdx.y * 32;
  int m = lane & 15, half = lane >> 4;
  v8f acc0 = vzero8(), acc1 = vzero8();
  int rA = tid >> 1, cbA = (tid & 1) * 16;
  int grA = m0 + rA;
  int arA = RSWAP ? ((grA % rowMod)*rowStride + grA / rowMod)
                  : ((grA / rowMod)*rowStride + grA % rowMod);
  const __bf16* srcA = A + (size_t)arA*lda + cbA;
  unsigned dstA = lds_addr32(&As[rA*32 + cbA]);
  for (int k0 = 0; k0 < K; k0 += 32) {
    // stage A 128x32 asynchronously (2 x 16B per thread)
    async_copy_b128(dstA,      srcA + k0);
    async_copy_b128(dstA + 16, srcA + k0 + 8);
    { // stage B 32x32 swizzled into tile-lane order
      int r = tid >> 3, c0 = (tid & 7) * 4;
      const __bf16* src = Bm + (size_t)(k0 + r)*ldb + n0 + c0;
      __bf16 h0 = src[0], h1 = src[1], h2 = src[2], h3 = src[3];
      int hk = r >> 4, ek = r & 15;
      int base = (hk << 4) | (c0 & 15);
      int nt = c0 >> 4;
      __bf16* dst = &Bs[(((nt << 5) + base) << 4) + ek];
      dst[0]  = h0;  dst[16] = h1;  dst[32] = h2;  dst[48] = h3;
    }
    async_wait0();
    __syncthreads();
    v8bf alo = *(const v8bf*)&As[(wave*16 + m)*32 + 16*half];
    v8bf ahi = *(const v8bf*)&As[(wave*16 + m)*32 + 16*half + 8];
    v16bf a = join16(alo, ahi);
    v16bf b0 = *(const v16bf*)&Bs[(0*32 + lane)*16];
    v16bf b1 = *(const v16bf*)&Bs[(1*32 + lane)*16];
    acc0 = wmma_bf16(a, b0, acc0);
    acc1 = wmma_bf16(a, b1, acc1);
    __syncthreads();
  }
#pragma unroll
  for (int v = 0; v < 8; ++v) {
    int gr = m0 + wave*16 + v + 8*half;
    int c0 = n0 + m, c1 = n0 + 16 + m;
    float v0 = acc0[v] + bias[c0];
    float v1 = acc1[v] + bias[c1];
    if (CBF) {
      ((__bf16*)C)[(size_t)gr*ldc + c0] = f2bf(v0);
      ((__bf16*)C)[(size_t)gr*ldc + c1] = f2bf(v1);
    } else {
      ((float*)C)[(size_t)gr*ldc + c0] = v0;
      ((float*)C)[(size_t)gr*ldc + c1] = v1;
    }
  }
}

// ---------------- BiLSTM scan ----------------
__global__ __launch_bounds__(256) void lstm_kernel(
    const float* __restrict__ xw_f, const float* __restrict__ xw_b,
    const __bf16* __restrict__ whhT_all,   // [2][32 tiles][32 lanes][16]
    __bf16* __restrict__ gat_in) {
  __shared__ __attribute__((aligned(32))) __bf16 wsh[32*32*16];
  __shared__ __attribute__((aligned(32))) __bf16 hbf[16*64];
  __shared__ float gact[16*256];
  __shared__ float csh[16*64];
  int dir = blockIdx.x;
  const float* xw = dir ? xw_b : xw_f;
  const __bf16* whhT = whhT_all + (size_t)dir*16384;
  int tid = threadIdx.x, wave = tid >> 5, lane = tid & 31;
  for (int i = tid*8; i < 16384; i += 256*8)
    async_copy_b128(lds_addr32(&wsh[i]), whhT + i);
  for (int i = tid; i < 16*64; i += 256) { hbf[i] = f2bf(0.f); csh[i] = 0.f; }
  async_wait0();
  __syncthreads();
  int m = lane & 15, half = lane >> 4;
  // this wave's 32 columns fall entirely inside one gate region; region 2 is tanh
  bool isTanh = (wave == 4 || wave == 5);
  for (int t = 0; t < Sc; ++t) {
    int idx = dir ? (Sc - 1 - t) : t;
    v8f acc0 = vzero8(), acc1 = vzero8();
#pragma unroll
    for (int ks = 0; ks < 2; ++ks) {
      v8bf alo = *(const v8bf*)&hbf[m*64 + ks*32 + 16*half];
      v8bf ahi = *(const v8bf*)&hbf[m*64 + ks*32 + 16*half + 8];
      v16bf a = join16(alo, ahi);
      v16bf b0 = *(const v16bf*)&wsh[((wave*4 + 0*2 + ks)*32 + lane)*16];
      v16bf b1 = *(const v16bf*)&wsh[((wave*4 + 1*2 + ks)*32 + lane)*16];
      acc0 = wmma_bf16(a, b0, acc0);
      acc1 = wmma_bf16(a, b1, acc1);
    }
#pragma unroll
    for (int v = 0; v < 8; ++v) {
      int bm = v + 8*half;
      int c0 = wave*32 + m, c1 = c0 + 16;
      float g0 = acc0[v] + xw[((size_t)idx*Bc + bm)*256 + c0];
      float g1 = acc1[v] + xw[((size_t)idx*Bc + bm)*256 + c1];
      gact[bm*256 + c0] = isTanh ? fast_tanh(g0) : fast_sigmoid(g0);
      gact[bm*256 + c1] = isTanh ? fast_tanh(g1) : fast_sigmoid(g1);
    }
    __syncthreads();
    {
      int b_ = tid >> 4;
      int j0 = (tid & 15) * 4;
#pragma unroll
      for (int q = 0; q < 4; ++q) {
        int j = j0 + q;
        float iv = gact[b_*256 + j];
        float fv = gact[b_*256 + 64 + j];
        float gv = gact[b_*256 + 128 + j];
        float ov = gact[b_*256 + 192 + j];
        float c = fv * csh[b_*64 + j] + iv * gv;
        float h = ov * fast_tanh(c);
        csh[b_*64 + j] = c;
        hbf[b_*64 + j] = f2bf(h);
        gat_in[((size_t)(b_*Nc + idx))*128 + dir*64 + j] = f2bf(h);
      }
    }
    __syncthreads();
  }
}

// ---------------- f1/f2 attention coefficients ----------------
__global__ void f1f2_kernel(const __bf16* __restrict__ H, int ldh, int Fsum,
                            const float* __restrict__ a, int aStride, int rowsPerHead,
                            float* __restrict__ f1, float* __restrict__ f2, int total) {
  int r = blockIdx.x * blockDim.x + threadIdx.x;
  if (r >= total) return;
  int head = r / rowsPerHead;
  const float* a1 = a + (size_t)head*aStride;
  const float* a2 = a1 + Fsum;
  const __bf16* hp = H + (size_t)r*ldh;
  float s1 = 0.f, s2 = 0.f;
  for (int f = 0; f < Fsum; ++f) {
    float hv = bf2f(hp[f]);
    s1 += hv * a1[f];
    s2 += hv * a2[f];
  }
  f1[r] = s1; f2[r] = s2;
}

// ---------------- fused masked-softmax GAT attention (flash-style) ----------------
template<int FT, int LDH>
__global__ __launch_bounds__(128) void attn_kernel(
    const __bf16* __restrict__ H, int headStride,
    const float* __restrict__ f1, const float* __restrict__ f2,
    const unsigned* __restrict__ adjbits,   // [B][N][AW]
    __bf16* outBf, int outLd,
    float* outF32,
    int blocksPerB) {
  int tid = threadIdx.x;
  int lane = tid & 31;
  int wv = tid >> 5;
  int head = blockIdx.y;
  int b   = blockIdx.x / blocksPerB;
  int blk = blockIdx.x % blocksPerB;
  int i0 = blk*64 + wv*16;
  const __bf16* Hh = H + (size_t)head*headStride*LDH;
  const float* f1h = f1 + (size_t)head*headStride + (size_t)b*Nc;
  const float* f2h = f2 + (size_t)head*headStride + (size_t)b*Nc;

  __shared__ __attribute__((aligned(32))) __bf16 vsw[FT*32*16];
  __shared__ float f2sh[Nc];

  for (int k = tid; k < Nc; k += 128)
    async_copy_b32(lds_addr32(&f2sh[k]), f2h + k);
  async_wait0();

  int m = lane & 15, half = lane >> 4;
  const float NINF = -__builtin_inff();
  float mrow = NINF, lrow = 0.f;
  float f1i = f1h[i0 + m];
  v8f acc[FT];
#pragma unroll
  for (int ft = 0; ft < FT; ++ft) acc[ft] = vzero8();

  const unsigned* adjrow = adjbits + ((size_t)(b*Nc + i0 + m))*AW;

  for (int j0 = 0; j0 < Nc; j0 += 32) {
    // stage V chunk (32 x LDH) swizzled into B-tile lane order
    for (int e0 = tid*16; e0 < 32*LDH; e0 += 128*16) {
      int r  = e0 / LDH;
      int c0 = e0 % LDH;
      const __bf16* src = Hh + (size_t)(b*Nc + j0 + r)*LDH + c0;
      v8bf u0 = *(const v8bf*)(src);
      v8bf u1 = *(const v8bf*)(src + 8);
      int hk = r >> 4, ek = r & 15;
#pragma unroll
      for (int q = 0; q < 8; ++q) {
        int c = c0 + q;
        vsw[((((c >> 4) << 5) + (hk << 4) + (c & 15)) << 4) + ek] = u0[q];
      }
#pragma unroll
      for (int q = 0; q < 8; ++q) {
        int c = c0 + 8 + q;
        vsw[((((c >> 4) << 5) + (hk << 4) + (c & 15)) << 4) + ek] = u1[q];
      }
    }
    __syncthreads();
    unsigned adjm = adjrow[j0 >> 5];
    float p[16];
    float cmax = NINF;
#pragma unroll
    for (int e = 0; e < 16; ++e) {
      int k = 16*half + e;
      float ev = NINF;
      if ((adjm >> k) & 1u) {
        float x = f1i + f2sh[j0 + k];
        ev = (x > 0.f) ? x : 0.2f * x;
        cmax = fmaxf(cmax, ev);
      }
      p[e] = ev;
    }
    cmax = fmaxf(cmax, __shfl_xor(cmax, 16, 32));
    float mnew = fmaxf(mrow, cmax);
    float scale = (mnew == NINF) ? 1.f : __expf(mrow - mnew);
    float rsum = 0.f;
#pragma unroll
    for (int e = 0; e < 16; ++e) {
      float pv = (p[e] == NINF) ? 0.f : __expf(p[e] - mnew);
      p[e] = pv; rsum += pv;
    }
    rsum += __shfl_xor(rsum, 16, 32);
    lrow = lrow * scale + rsum;
    mrow = mnew;
    float sc[8];
#pragma unroll
    for (int v = 0; v < 8; ++v) sc[v] = __shfl(scale, v + 8*half, 32);
    v16bf a;
#pragma unroll
    for (int e = 0; e < 16; ++e) a[e] = f2bf(p[e]);
#pragma unroll
    for (int ft = 0; ft < FT; ++ft) {
#pragma unroll
      for (int v = 0; v < 8; ++v) acc[ft][v] *= sc[v];
      v16bf bv = *(const v16bf*)&vsw[((ft << 5) + lane)*16];
      acc[ft] = wmma_bf16(a, bv, acc[ft]);
    }
    __syncthreads();
  }
  float ldiv[8];
#pragma unroll
  for (int v = 0; v < 8; ++v) ldiv[v] = __shfl(lrow, v + 8*half, 32);
#pragma unroll
  for (int ft = 0; ft < FT; ++ft) {
#pragma unroll
    for (int v = 0; v < 8; ++v) {
      int row = v + 8*half;
      int col = ft*16 + m;
      float val = acc[ft][v] * fast_rcp(ldiv[v]);
      val = (val > 0.f) ? val : (__expf(val) - 1.f);  // elu
      int gi = i0 + row;
      if (outBf) {
        outBf[(size_t)(b*Nc + gi)*outLd + head*FT*16 + col] = f2bf(val);
      } else if (col < Tc) {
        outF32[((size_t)b*Sc + gi)*Tc + col] = val;
      }
    }
  }
}

// ---------------- fuse ----------------
__global__ void fuse_kernel(const float* __restrict__ proj,
                            const float* __restrict__ g0, const float* __restrict__ g1,
                            const float* __restrict__ g2, const float* __restrict__ fw,
                            float* __restrict__ feats, int total) {
  int i = blockIdx.x * blockDim.x + threadIdx.x;
  if (i >= total) return;
  int r = i / Tc, t = i % Tc;
  feats[i] = fw[0]*proj[(size_t)r*32 + t] + fw[1]*g0[i] + fw[2]*g1[i] + fw[3]*g2[i];
}

// ---------------- Viterbi: one wave per batch ----------------
__global__ __launch_bounds__(32) void viterbi_kernel(const float* __restrict__ feats,
                                                     const float* __restrict__ trans,
                                                     int* __restrict__ bp,
                                                     float* __restrict__ out) {
  int b = blockIdx.x, lane = threadIdx.x;
  __shared__ float psh[32];
  const float NINF = -__builtin_inff();
  float part = NINF;
  if (lane < Tc) part = feats[((size_t)b*Sc)*Tc + lane] + trans[STARTc*Tc + lane];
  for (int s = 1; s < Sc; ++s) {
    psh[lane] = part;
    __syncthreads();
    float best = NINF; int bpi = 0;
    if (lane < Tc) {
      float ft = feats[((size_t)b*Sc + s)*Tc + lane];
      for (int p = 0; p < Tc; ++p) {
        float v = psh[p] + trans[p*Tc + lane] + ft;
        if (v > best) { best = v; bpi = p; }
      }
      bp[((size_t)b*(Sc-1) + (s-1))*Tc + lane] = bpi;
    }
    __syncthreads();
    part = best;
  }
  psh[lane] = (lane < Tc) ? (part + trans[lane*Tc + STOPc]) : NINF;
  __syncthreads();
  if (lane == 0) {
    int tag = 0; float bv = NINF;
    for (int c = 0; c < Tc; ++c) if (psh[c] > bv) { bv = psh[c]; tag = c; }
    out[(size_t)b*Sc + Sc - 1] = (float)tag;
    for (int s = Sc - 1; s >= 1; --s) {
      tag = bp[((size_t)b*(Sc-1) + (s-1))*Tc + tag];
      out[(size_t)b*Sc + s - 1] = (float)tag;
    }
  }
}

// ---------------- host ----------------
extern "C" void kernel_launch(void* const* d_in, const int* in_sizes, int n_in,
                              void* d_out, int out_size, void* d_ws, size_t ws_size,
                              hipStream_t stream) {
  (void)in_sizes; (void)n_in; (void)out_size; (void)ws_size;
  const int*   batch_char = (const int*)  d_in[0];
  const int*   gaz_list   = (const int*)  d_in[2];
  const int*   g_t = (const int*)d_in[3];
  const int*   g_c = (const int*)d_in[4];
  const int*   g_l = (const int*)d_in[5];
  const float* char_table = (const float*)d_in[7];
  const float* gaz_table  = (const float*)d_in[8];
  const float* w_ih_f = (const float*)d_in[9];
  const float* w_hh_f = (const float*)d_in[10];
  const float* b_f    = (const float*)d_in[11];
  const float* w_ih_b = (const float*)d_in[12];
  const float* w_hh_b = (const float*)d_in[13];
  const float* b_b    = (const float*)d_in[14];
  const float* h2h_W  = (const float*)d_in[15];
  const float* h2h_b  = (const float*)d_in[16];
  const float* gat_Wh = (const float*)d_in[17];
  const float* gat_ah = (const float*)d_in[18];
  const float* gat_Wo = (const float*)d_in[19];
  const float* gat_ao = (const float*)d_in[20];
  const float* fuse_w = (const float*)d_in[21];
  const float* transitions = (const float*)d_in[22];
  float* out = (float*)d_out;

  const size_t BS = (size_t)Bc*Sc;   // 6144
  const size_t BN = (size_t)Bc*Nc;   // 12288
  const int wordsPerGraph = Bc*Nc*AW;   // 294912

  size_t off = 0;
  auto alc = [&](size_t bytes) { off = (off + 255) & ~(size_t)255; size_t o = off; off += bytes; return o; };
  char* wsp = (char*)d_ws;
  __bf16*   emb     = (__bf16*)  (wsp + alc(BS*128*2));
  __bf16*   wihT    = (__bf16*)  (wsp + alc(2*128*256*2));
  __bf16*   whhT    = (__bf16*)  (wsp + alc(2*64*256*2));
  __bf16*   WhT     = (__bf16*)  (wsp + alc(12*128*64*2));
  __bf16*   WoT     = (__bf16*)  (wsp + alc(3*256*32*2));
  __bf16*   h2hT    = (__bf16*)  (wsp + alc(128*32*2));
  float*    h2hbP   = (float*)   (wsp + alc(32*4));
  float*    zbias   = (float*)   (wsp + alc(256*4));
  float*    xw      = (float*)   (wsp + alc(2*BS*256*4));   // [dir][S][B][256]
  __bf16*   gat_in  = (__bf16*)  (wsp + alc(BN*128*2));
  unsigned* adjbits = (unsigned*)(wsp + alc((size_t)3*wordsPerGraph*4));
  __bf16*   hfeat   = (__bf16*)  (wsp + alc(4*BN*64*2));    // per-graph reuse
  float*    f1      = (float*)   (wsp + alc(4*BN*4));
  float*    f2      = (float*)   (wsp + alc(4*BN*4));
  __bf16*   hcat    = (__bf16*)  (wsp + alc(BN*256*2));
  __bf16*   ho      = (__bf16*)  (wsp + alc(BN*32*2));
  float*    f1o     = (float*)   (wsp + alc(BN*4));
  float*    f2o     = (float*)   (wsp + alc(BN*4));
  float*    gout    = (float*)   (wsp + alc(3*BS*Tc*4));
  float*    proj    = (float*)   (wsp + alc(BS*32*4));
  float*    feats   = (float*)   (wsp + alc(BS*Tc*4));
  int*      bp      = (int*)     (wsp + alc((size_t)Bc*(Sc-1)*Tc*4));

  const int IDMOD = 1 << 30;

  // 1) weight prep + zero bias
  {
    int total = 2*128*256 + 2*64*256 + 12*128*64 + 3*256*32 + 128*32 + 32 + 256;
    prep_kernel<<<(total + 255)/256, 256, 0, stream>>>(
        w_ih_f, w_ih_b, w_hh_f, w_hh_b, h2h_W, h2h_b, gat_Wh, gat_Wo,
        wihT, whhT, WhT, WoT, h2hT, h2hbP, zbias);
  }
  // 2) adjacency bit-pack
  packadj_kernel<<<(3*wordsPerGraph + 255)/256, 256, 0, stream>>>(
      g_t, g_c, g_l, adjbits, wordsPerGraph);
  // 3) gathers
  embed_kernel<<<(int)(BS*128/256), 256, 0, stream>>>(batch_char, char_table, emb, (int)(BS*128));
  gaz_kernel<<<(int)((size_t)Bc*Gc*128/256), 256, 0, stream>>>(gaz_list, gaz_table, gat_in,
                                                               (int)((size_t)Bc*Gc*128));
  // 4) xw = emb @ w_ih^T + b (both dirs), output rows re-ordered to [S][B]
  for (int dir = 0; dir < 2; ++dir) {
    gemm_kernel<false, true><<<dim3(48, 8), 256, 0, stream>>>(
        emb, 128, 16, Sc, wihT + (size_t)dir*128*256, 256,
        xw + (size_t)dir*BS*256, 256, dir ? b_b : b_f, 128);
  }
  // 5) BiLSTM scan
  lstm_kernel<<<2, 256, 0, stream>>>(xw, xw + BS*256, whhT, gat_in);
  // 6) lstm_proj : rows remapped b*768+s
  gemm_kernel<false, false><<<dim3(48, 1), 256, 0, stream>>>(
      gat_in, 128, Sc, Nc, h2hT, 32, proj, 32, h2hbP, 128);
  // 7) per-graph GAT
  for (int g = 0; g < 3; ++g) {
    const unsigned* abits = adjbits + (size_t)g*wordsPerGraph;
    for (int h = 0; h < 4; ++h) {
      gemm_kernel<true, false><<<dim3(96, 2), 256, 0, stream>>>(
          gat_in, 128, IDMOD, 0, WhT + (size_t)(g*4 + h)*128*64, 64,
          hfeat + (size_t)h*BN*64, 64, zbias, 128);
    }
    f1f2_kernel<<<(int)(4*BN/256), 256, 0, stream>>>(
        hfeat, 64, 64, gat_ah + (size_t)g*4*128, 128, (int)BN, f1, f2, (int)(4*BN));
    attn_kernel<4, 64><<<dim3(Bc*12, 4), 128, 0, stream>>>(
        hfeat, (int)BN, f1, f2, abits, hcat, 256, nullptr, 12);
    gemm_kernel<true, false><<<dim3(96, 1), 256, 0, stream>>>(
        hcat, 256, IDMOD, 0, WoT + (size_t)g*256*32, 32, ho, 32, zbias, 256);
    f1f2_kernel<<<(int)(BN/256), 256, 0, stream>>>(
        ho, 32, Tc, gat_ao + (size_t)g*2*Tc, 0, (int)BN, f1o, f2o, (int)BN);
    attn_kernel<2, 32><<<dim3(Bc*6, 1), 128, 0, stream>>>(
        ho, 0, f1o, f2o, abits, nullptr, 0, gout + (size_t)g*BS*Tc, 6);
  }
  // 8) fuse
  fuse_kernel<<<(int)((BS*Tc + 255)/256), 256, 0, stream>>>(
      proj, gout, gout + BS*Tc, gout + 2*BS*Tc, fuse_w, feats, (int)(BS*Tc));
  // 9) Viterbi decode
  viterbi_kernel<<<Bc, 32, 0, stream>>>(feats, transitions, bp, out);
}
